// IterativeRefinementCTC_29609504539066
// MI455X (gfx1250) — matrix operand order repaired and checked by reading, not compiled, not run on hardware
//
#include <hip/hip_runtime.h>

// ---------------- constants ----------------
#define BB   4
#define T    1024
#define DD   1024
#define H    16
#define HD   64
#define S2   64          // 2*SPAN
#define MAXREL 128

typedef __attribute__((ext_vector_type(16))) __bf16          v16bf;
typedef __attribute__((ext_vector_type(16))) unsigned short  v16u;
typedef __attribute__((ext_vector_type(8)))  float           v8f;

// ---------------- helpers ----------------
__device__ inline unsigned short f2bf(float f) {
  unsigned int u = __builtin_bit_cast(unsigned int, f);
  unsigned int r = u + 0x7FFFu + ((u >> 16) & 1u);
  if ((u & 0x7FFFFFFFu) > 0x7F800000u) r = u | 0x00400000u;  // quiet NaN
  return (unsigned short)(r >> 16);
}

__device__ inline v8f wmma_bf16(v16bf a, v16bf b, v8f c) {
  return __builtin_amdgcn_wmma_f32_16x16x32_bf16(false, a, false, b, (short)0, c,
                                                 false, false);
}

// A fragment, 16x32 bf16, row-major source with row stride lda.
// ISA 7.12.2: lane<16: row=lane, K=0..7 (v0-3) & 16..23 (v4-7);
//             lane>=16: same rows, K=8..15 & 24..31.
__device__ inline v16bf load_A_frag(const unsigned short* base, int lda, int kb,
                                    int lane) {
  int hl = lane & 15, hi = lane >> 4;
  const unsigned short* p = base + (size_t)hl * lda + kb;
  v16u t;
#pragma unroll
  for (int v = 0; v < 4; ++v) {
    int k0 = 2 * v + hi * 8;
    t[2 * v]     = p[k0];
    t[2 * v + 1] = p[k0 + 1];
  }
#pragma unroll
  for (int v = 4; v < 8; ++v) {
    int k0 = 16 + 2 * (v - 4) + hi * 8;
    t[2 * v]     = p[k0];
    t[2 * v + 1] = p[k0 + 1];
  }
  return __builtin_bit_cast(v16bf, t);
}

// B fragment, 32x16 bf16 (KxN). Source layout: column n contiguous over K
// (i.e. B^T row-major with row stride colStride).
// lane<16: col=lane, K=0..7 & 16..23; lane>=16: col=lane-16, K=8..15 & 24..31.
__device__ inline v16bf load_B_frag(const unsigned short* base, int colStride,
                                    int kb, int lane) {
  int hl = lane & 15, hi = lane >> 4;
  const unsigned short* p = base + (size_t)hl * colStride + kb + hi * 8;
  v16u t;
#pragma unroll
  for (int j = 0; j < 16; ++j) t[j] = p[j < 8 ? j : j + 8];
  return __builtin_bit_cast(v16bf, t);
}

// ---------------- simple conversion kernels ----------------
__global__ __launch_bounds__(256) void cvt_bf16_k(const float* __restrict__ s,
                                                  unsigned short* __restrict__ d,
                                                  int n) {
  int i = (blockIdx.x * 256 + threadIdx.x) * 4;
  if (i + 3 < n) {
    float4 f = *(const float4*)(s + i);
    d[i] = f2bf(f.x); d[i + 1] = f2bf(f.y);
    d[i + 2] = f2bf(f.z); d[i + 3] = f2bf(f.w);
  } else {
    for (int j = i; j < n; ++j) d[j] = f2bf(s[j]);
  }
}

// W[k][n] (1024x1024 f32) -> Wt[n][k] bf16
__global__ __launch_bounds__(256) void transpose_bf16_k(
    const float* __restrict__ src, unsigned short* __restrict__ dst) {
  __shared__ float tile[32][33];
  int tx = threadIdx.x, ty = threadIdx.y;  // 32 x 8
  int x = blockIdx.x * 32 + tx;
  int y0 = blockIdx.y * 32;
#pragma unroll
  for (int j = 0; j < 32; j += 8)
    tile[ty + j][tx] = src[(size_t)(y0 + ty + j) * DD + x];
  __syncthreads();
  int xo = blockIdx.y * 32 + tx;
  int y1 = blockIdx.x * 32;
#pragma unroll
  for (int j = 0; j < 32; j += 8)
    dst[(size_t)(y1 + ty + j) * DD + xo] = f2bf(tile[tx][ty + j]);
}

// LayerNorm rows of rel_emb [64][1024] -> bf16
__global__ __launch_bounds__(256) void layernorm_k(
    const float* __restrict__ x, const float* __restrict__ g,
    const float* __restrict__ bet, unsigned short* __restrict__ out) {
  __shared__ float red[256];
  int tid = threadIdx.x;
  int row = blockIdx.x;
  const float* xr = x + (size_t)row * DD;
  float s = 0.f;
  for (int i = tid; i < DD; i += 256) s += xr[i];
  red[tid] = s; __syncthreads();
  for (int st = 128; st > 0; st >>= 1) {
    if (tid < st) red[tid] += red[tid + st];
    __syncthreads();
  }
  float mu = red[0] * (1.0f / DD); __syncthreads();
  float v = 0.f;
  for (int i = tid; i < DD; i += 256) { float d = xr[i] - mu; v += d * d; }
  red[tid] = v; __syncthreads();
  for (int st = 128; st > 0; st >>= 1) {
    if (tid < st) red[tid] += red[tid + st];
    __syncthreads();
  }
  float rstd = rsqrtf(red[0] * (1.0f / DD) + 1e-5f);
  for (int i = tid; i < DD; i += 256)
    out[(size_t)row * DD + i] = f2bf((xr[i] - mu) * rstd * g[i] + bet[i]);
}

// ---------------- generic WMMA GEMM (ping-pong pipelined K loop) ----------------
// C[m,n] = sum_k A[m,k]*Bt[n,k] + bias[n]     (K must be a multiple of 64)
// mode 0: bf16 row-major; 1: f32 row-major (final out);
// mode 2: bf16 V^T head layout [(b*H+h)*HD+hd][T]; mode 3: bf16 pos [(h*S2+m)][hd]
__global__ __launch_bounds__(128) void gemm_bf16_k(
    const unsigned short* __restrict__ A, int lda,
    const unsigned short* __restrict__ Bt, int ldbt,
    const float* __restrict__ bias, void* __restrict__ Cout, int N, int K,
    int mode) {
  int tid = threadIdx.x, w = tid >> 5, lane = tid & 31;
  int hl = lane & 15, hi = lane >> 4;
  int m0 = blockIdx.x * 64 + w * 16;
  int n0 = blockIdx.y * 64;
  v8f z = {0.f, 0.f, 0.f, 0.f, 0.f, 0.f, 0.f, 0.f};
  v8f acc[4] = {z, z, z, z};
  const unsigned short* Abase = A + (size_t)m0 * lda;

  // two in-flight 32-K chunks; each register set is written by its own loads
  // and consumed by its own WMMAs -> no rotation copies.
  v16bf afA = load_A_frag(Abase, lda, 0, lane);
  v16bf bfA[4];
#pragma unroll
  for (int nt = 0; nt < 4; ++nt)
    bfA[nt] = load_B_frag(Bt + (size_t)(n0 + nt * 16) * ldbt, ldbt, 0, lane);
  v16bf afB = load_A_frag(Abase, lda, 32, lane);
  v16bf bfB[4];
#pragma unroll
  for (int nt = 0; nt < 4; ++nt)
    bfB[nt] = load_B_frag(Bt + (size_t)(n0 + nt * 16) * ldbt, ldbt, 32, lane);

  for (int kb = 64; kb < K; kb += 64) {
#pragma unroll
    for (int nt = 0; nt < 4; ++nt) acc[nt] = wmma_bf16(afA, bfA[nt], acc[nt]);
    afA = load_A_frag(Abase, lda, kb, lane);
#pragma unroll
    for (int nt = 0; nt < 4; ++nt)
      bfA[nt] = load_B_frag(Bt + (size_t)(n0 + nt * 16) * ldbt, ldbt, kb, lane);
#pragma unroll
    for (int nt = 0; nt < 4; ++nt) acc[nt] = wmma_bf16(afB, bfB[nt], acc[nt]);
    afB = load_A_frag(Abase, lda, kb + 32, lane);
#pragma unroll
    for (int nt = 0; nt < 4; ++nt)
      bfB[nt] =
          load_B_frag(Bt + (size_t)(n0 + nt * 16) * ldbt, ldbt, kb + 32, lane);
  }
#pragma unroll
  for (int nt = 0; nt < 4; ++nt) acc[nt] = wmma_bf16(afA, bfA[nt], acc[nt]);
#pragma unroll
  for (int nt = 0; nt < 4; ++nt) acc[nt] = wmma_bf16(afB, bfB[nt], acc[nt]);

#pragma unroll
  for (int nt = 0; nt < 4; ++nt) {
    int n = n0 + nt * 16 + hl;
    float bv = bias ? bias[n] : 0.f;
#pragma unroll
    for (int r = 0; r < 8; ++r) {
      int m = m0 + r + 8 * hi;
      float val = acc[nt][r] + bv;
      if (mode == 0) {
        ((unsigned short*)Cout)[(size_t)m * N + n] = f2bf(val);
      } else if (mode == 1) {
        ((float*)Cout)[(size_t)m * N + n] = val;
      } else if (mode == 2) {  // vT: m->(b,t), n->(h,hd)
        int b_ = m >> 10, t_ = m & 1023, h_ = n >> 6, hd_ = n & 63;
        ((unsigned short*)Cout)[((size_t)(b_ * H + h_) * HD + hd_) * T + t_] =
            f2bf(val);
      } else {  // pos: [(h*S2 + m)*HD + hd]
        int h_ = n >> 6, hd_ = n & 63;
        ((unsigned short*)Cout)[((size_t)h_ * S2 + m) * HD + hd_] = f2bf(val);
      }
    }
  }
}

// ---------------- c2p / p2c projection GEMM ----------------
// out[bh][t][s] = scale * sum_d act[b,t,(h,d)] * pos[h][s][d]
__global__ __launch_bounds__(128) void posbias_k(
    const unsigned short* __restrict__ act, const unsigned short* __restrict__ pos,
    float* __restrict__ out, float scale) {
  int tid = threadIdx.x, w = tid >> 5, lane = tid & 31;
  int hl = lane & 15, hi = lane >> 4;
  int bh = blockIdx.y, b = bh >> 4, h = bh & 15;
  int t0 = blockIdx.x * 64 + w * 16;
  const unsigned short* Abase = act + ((size_t)(b * T + t0)) * DD + h * HD;
  const unsigned short* Pbase = pos + (size_t)h * S2 * HD;
  v8f z = {0.f, 0.f, 0.f, 0.f, 0.f, 0.f, 0.f, 0.f};
  v8f acc[4] = {z, z, z, z};
  // batch all fragment loads, then all WMMAs
  v16bf af0 = load_A_frag(Abase, DD, 0, lane);
  v16bf af1 = load_A_frag(Abase, DD, 32, lane);
  v16bf bf[8];
#pragma unroll
  for (int nt = 0; nt < 4; ++nt) {
    bf[2 * nt]     = load_B_frag(Pbase + (size_t)nt * 16 * HD, HD, 0, lane);
    bf[2 * nt + 1] = load_B_frag(Pbase + (size_t)nt * 16 * HD, HD, 32, lane);
  }
#pragma unroll
  for (int nt = 0; nt < 4; ++nt) {
    acc[nt] = wmma_bf16(af0, bf[2 * nt], acc[nt]);
    acc[nt] = wmma_bf16(af1, bf[2 * nt + 1], acc[nt]);
  }
#pragma unroll
  for (int nt = 0; nt < 4; ++nt)
#pragma unroll
    for (int r = 0; r < 8; ++r)
      out[((size_t)bh * T + t0 + r + 8 * hi) * S2 + nt * 16 + hl] =
          acc[nt][r] * scale;
}

// ---------------- log-bucket LUTs ----------------
__device__ inline int bucket_pos(int rel) {
  const int mid = 16;  // SPAN/2
  int sg = (rel > 0) ? 1 : ((rel < 0) ? -1 : 0);
  int ap = (rel < mid && rel > -mid) ? (mid - 1) : (rel < 0 ? -rel : rel);
  float logp = ceilf(logf((float)ap / (float)mid) /
                     logf((float)(MAXREL - 1) / (float)mid) * (float)(mid - 1)) +
               (float)mid;
  return (ap <= mid) ? rel : (int)(logp * (float)sg);
}

__global__ __launch_bounds__(256) void lut_k(unsigned char* __restrict__ lc,
                                             unsigned char* __restrict__ lp) {
  int i = blockIdx.x * 256 + threadIdx.x;
  if (i >= 2 * T - 1) return;
  int delta = i - (T - 1);                 // q - k
  int ci = bucket_pos(delta) + 32;         // clip(bucket(q-k)+SPAN)
  int pi = -bucket_pos(-delta) + 32;       // clip(-bucket(k-q)+SPAN)
  ci = ci < 0 ? 0 : (ci > 63 ? 63 : ci);
  pi = pi < 0 ? 0 : (pi > 63 ? 63 : pi);
  lc[i] = (unsigned char)ci;
  lp[i] = (unsigned char)pi;
}

// ---------------- fused disentangled flash attention ----------------
__global__ __launch_bounds__(128) void attn_k(
    const unsigned short* __restrict__ q_bf, const unsigned short* __restrict__ k_bf,
    const unsigned short* __restrict__ vT_bf, const float* __restrict__ c2p_full,
    const float* __restrict__ p2c_full, const unsigned char* __restrict__ lutc,
    const unsigned char* __restrict__ lutp, const unsigned char* __restrict__ mask,
    unsigned short* __restrict__ ctx_bf) {
  __shared__ float c2p_s[64][S2];          // 16 KB: q-tile rows of c2p
  __shared__ float p2c_s[64][S2];          // 16 KB: key-tile rows of p2c
  __shared__ float msk_s[64];
  __shared__ unsigned char lc_s[2048], lp_s[2048];
  __shared__ unsigned short p_s[4][16][64];  // per-wave prob relayout (8 KB)

  int tid = threadIdx.x, w = tid >> 5, lane = tid & 31;
  int hl = lane & 15, hi = lane >> 4;
  int bh = blockIdx.y, b = bh >> 4, h = bh & 15;
  int qt = blockIdx.x * 64;

  for (int i = tid; i < 2 * T - 1; i += 128) { lc_s[i] = lutc[i]; lp_s[i] = lutp[i]; }
  for (int i = tid; i < 64 * S2; i += 128) {
    int r = i >> 6, c = i & 63;
    c2p_s[r][c] = c2p_full[((size_t)bh * T + qt + r) * S2 + c];
  }

  const unsigned short* qbase = q_bf + ((size_t)(b * T + qt + w * 16)) * DD + h * HD;
  v16bf qf0 = load_A_frag(qbase, DD, 0, lane);
  v16bf qf1 = load_A_frag(qbase, DD, 32, lane);

  v8f z = {0.f, 0.f, 0.f, 0.f, 0.f, 0.f, 0.f, 0.f};
  v8f oacc[4] = {z, z, z, z};
  float mrow[8], lrow[8];
#pragma unroll
  for (int r = 0; r < 8; ++r) { mrow[r] = -INFINITY; lrow[r] = 0.f; }
  const float inv_scale = 0.07216878364870323f;  // 1/sqrt(64*3)

  for (int kt = 0; kt < T; kt += 64) {
    __syncthreads();  // protect p2c_s/msk_s readers (and initial staging)
    for (int i = tid; i < 64 * S2; i += 128) {
      int r = i >> 6, c = i & 63;
      p2c_s[r][c] = p2c_full[((size_t)bh * T + kt + r) * S2 + c];
    }
    if (tid < 64) msk_s[tid] = mask[b * T + kt + tid] ? -INFINITY : 0.0f;
    __syncthreads();

    // gfx1250 prefetch of next key tile's K/V (L2-targeted)
    if (kt + 64 < T) {
      int pr = tid & 63;
      if (tid < 64) {
        __builtin_prefetch(
            k_bf + ((size_t)(b * T + kt + 64 + pr)) * DD + h * HD, 0, 3);
      } else {
        __builtin_prefetch(
            vT_bf + ((size_t)(bh * HD + pr)) * T + kt + 64, 0, 3);
      }
    }

    // ---- scores: q @ k^T (batch all loads, then all WMMAs) ----
    v16bf kf[8];
#pragma unroll
    for (int nt = 0; nt < 4; ++nt) {
      const unsigned short* kbase =
          k_bf + ((size_t)(b * T + kt + nt * 16)) * DD + h * HD;
      kf[2 * nt]     = load_B_frag(kbase, DD, 0, lane);
      kf[2 * nt + 1] = load_B_frag(kbase, DD, 32, lane);
    }
    v8f sa[4];
#pragma unroll
    for (int nt = 0; nt < 4; ++nt) {
      sa[nt] = z;
      sa[nt] = wmma_bf16(qf0, kf[2 * nt], sa[nt]);
      sa[nt] = wmma_bf16(qf1, kf[2 * nt + 1], sa[nt]);
    }

    // issue V-fragment loads now; softmax VALU below hides their latency
    v16bf vf[8];
#pragma unroll
    for (int nt = 0; nt < 4; ++nt) {
      const unsigned short* vb = vT_bf + ((size_t)(bh * HD + nt * 16)) * T + kt;
      vf[2 * nt]     = load_B_frag(vb, T, 0, lane);
      vf[2 * nt + 1] = load_B_frag(vb, T, 32, lane);
    }

    // ---- bias add ----
    float s[4][8];
#pragma unroll
    for (int nt = 0; nt < 4; ++nt) {
      int kl = nt * 16 + hl;
      int kg = kt + kl;
      float mk = msk_s[kl];
#pragma unroll
      for (int r = 0; r < 8; ++r) {
        int qrow = w * 16 + r + 8 * hi;
        int d = (qt + qrow) - kg + (T - 1);
        float bias = c2p_s[qrow][lc_s[d]] + p2c_s[kl][lp_s[d]];
        s[nt][r] = sa[nt][r] * inv_scale + bias + mk;
      }
    }

    // ---- online softmax (rows r+8*hi, 64 cols across 4 tiles x 16 lanes) ----
#pragma unroll
    for (int r = 0; r < 8; ++r) {
      float mx = fmaxf(fmaxf(s[0][r], s[1][r]), fmaxf(s[2][r], s[3][r]));
#pragma unroll
      for (int off = 8; off >= 1; off >>= 1) mx = fmaxf(mx, __shfl_xor(mx, off, 32));
      float mnew = fmaxf(mrow[r], mx);
      float corr = (mrow[r] == -INFINITY) ? 0.0f : __expf(mrow[r] - mnew);
      float rsum = 0.f;
#pragma unroll
      for (int nt = 0; nt < 4; ++nt) {
        float p = (s[nt][r] == -INFINITY) ? 0.0f : __expf(s[nt][r] - mnew);
        s[nt][r] = p;
        rsum += p;
      }
#pragma unroll
      for (int off = 8; off >= 1; off >>= 1) rsum += __shfl_xor(rsum, off, 32);
      lrow[r] = lrow[r] * corr + rsum;
      mrow[r] = mnew;
#pragma unroll
      for (int nt = 0; nt < 4; ++nt) oacc[nt][r] = oacc[nt][r] * corr;
    }

    // ---- relayout probs C-frag -> A-frag via per-wave LDS ----
#pragma unroll
    for (int nt = 0; nt < 4; ++nt)
#pragma unroll
      for (int r = 0; r < 8; ++r)
        p_s[w][r + 8 * hi][nt * 16 + hl] = f2bf(s[nt][r]);
    // same-wave RAW; compiler inserts s_wait_dscnt
    v16bf pf0 = load_A_frag(&p_s[w][0][0], 64, 0, lane);
    v16bf pf1 = load_A_frag(&p_s[w][0][0], 64, 32, lane);

    // ---- probs @ V ----
#pragma unroll
    for (int nt = 0; nt < 4; ++nt) {
      oacc[nt] = wmma_bf16(pf0, vf[2 * nt], oacc[nt]);
      oacc[nt] = wmma_bf16(pf1, vf[2 * nt + 1], oacc[nt]);
    }
  }

  // ---- epilogue: ctx[b, t, h*HD+hd] bf16 ----
#pragma unroll
  for (int nt = 0; nt < 4; ++nt)
#pragma unroll
    for (int r = 0; r < 8; ++r) {
      int qrow = w * 16 + r + 8 * hi;
      float val = oacc[nt][r] / lrow[r];
      ctx_bf[((size_t)(b * T + qt + qrow)) * DD + h * HD + nt * 16 + hl] = f2bf(val);
    }
}

// ---------------- host launcher ----------------
extern "C" void kernel_launch(void* const* d_in, const int* in_sizes, int n_in,
                              void* d_out, int out_size, void* d_ws, size_t ws_size,
                              hipStream_t stream) {
  const float* Xq   = (const float*)d_in[0];
  const float* Xkv  = (const float*)d_in[1];
  const unsigned char* mask = (const unsigned char*)d_in[2];
  const float* Wq = (const float*)d_in[3];  const float* bq = (const float*)d_in[4];
  const float* Wk = (const float*)d_in[5];  const float* bk = (const float*)d_in[6];
  const float* Wv = (const float*)d_in[7];  const float* bv = (const float*)d_in[8];
  const float* Wo = (const float*)d_in[9];  const float* bo = (const float*)d_in[10];
  const float* rel_emb = (const float*)d_in[11];
  const float* ln_g = (const float*)d_in[12]; const float* ln_b = (const float*)d_in[13];
  const float* Wpk = (const float*)d_in[14]; const float* bpk = (const float*)d_in[15];
  const float* Wpq = (const float*)d_in[16]; const float* bpq = (const float*)d_in[17];

  char* wsp = (char*)d_ws;
  size_t off = 0;
  auto alloc = [&](size_t bytes) {
    void* p = wsp + off;
    off += (bytes + 255) & ~(size_t)255;
    return p;
  };
  const size_t actE = (size_t)BB * T * DD;          // 4M elements
  unsigned short* Xq_bf  = (unsigned short*)alloc(actE * 2);
  unsigned short* Xkv_bf = (unsigned short*)alloc(actE * 2);
  unsigned short* WqT = (unsigned short*)alloc((size_t)DD * DD * 2);
  unsigned short* WkT = (unsigned short*)alloc((size_t)DD * DD * 2);
  unsigned short* WvT = (unsigned short*)alloc((size_t)DD * DD * 2);
  unsigned short* WoT = (unsigned short*)alloc((size_t)DD * DD * 2);
  unsigned short* WpkT = (unsigned short*)alloc((size_t)DD * DD * 2);
  unsigned short* WpqT = (unsigned short*)alloc((size_t)DD * DD * 2);
  unsigned short* relln = (unsigned short*)alloc((size_t)S2 * DD * 2);
  unsigned short* posk_bf = (unsigned short*)alloc((size_t)H * S2 * HD * 2);
  unsigned short* posq_bf = (unsigned short*)alloc((size_t)H * S2 * HD * 2);
  unsigned short* q_bf  = (unsigned short*)alloc(actE * 2);
  unsigned short* k_bf  = (unsigned short*)alloc(actE * 2);
  unsigned short* vT_bf = (unsigned short*)alloc(actE * 2);
  float* c2p_full = (float*)alloc((size_t)BB * H * T * S2 * 4);
  float* p2c_full = (float*)alloc((size_t)BB * H * T * S2 * 4);
  unsigned short* ctx_bf = (unsigned short*)alloc(actE * 2);
  unsigned char* lutc = (unsigned char*)alloc(2048);
  unsigned char* lutp = (unsigned char*)alloc(2048);

  // 1) conversions
  cvt_bf16_k<<<(int)(actE / 1024), 256, 0, stream>>>(Xq, Xq_bf, (int)actE);
  cvt_bf16_k<<<(int)(actE / 1024), 256, 0, stream>>>(Xkv, Xkv_bf, (int)actE);
  dim3 tg(32, 32), tb(32, 8);
  transpose_bf16_k<<<tg, tb, 0, stream>>>(Wq, WqT);
  transpose_bf16_k<<<tg, tb, 0, stream>>>(Wk, WkT);
  transpose_bf16_k<<<tg, tb, 0, stream>>>(Wv, WvT);
  transpose_bf16_k<<<tg, tb, 0, stream>>>(Wo, WoT);
  transpose_bf16_k<<<tg, tb, 0, stream>>>(Wpk, WpkT);
  transpose_bf16_k<<<tg, tb, 0, stream>>>(Wpq, WpqT);

  // 2) rel layernorm
  layernorm_k<<<S2, 256, 0, stream>>>(rel_emb, ln_g, ln_b, relln);

  // 3) projections (WMMA GEMMs)
  dim3 gBig(BB * T / 64, DD / 64);
  gemm_bf16_k<<<gBig, 128, 0, stream>>>(Xq_bf, DD, WqT, DD, bq, q_bf, DD, DD, 0);
  gemm_bf16_k<<<gBig, 128, 0, stream>>>(Xkv_bf, DD, WkT, DD, bk, k_bf, DD, DD, 0);
  gemm_bf16_k<<<gBig, 128, 0, stream>>>(Xkv_bf, DD, WvT, DD, bv, vT_bf, DD, DD, 2);
  dim3 gPos(1, DD / 64);
  gemm_bf16_k<<<gPos, 128, 0, stream>>>(relln, DD, WpkT, DD, bpk, posk_bf, DD, DD, 3);
  gemm_bf16_k<<<gPos, 128, 0, stream>>>(relln, DD, WpqT, DD, bpq, posq_bf, DD, DD, 3);

  // 4) c2p / p2c tables (1/sqrt(2*HD) folded in)
  const float inv_bias_scale = 0.08838834764831845f;
  dim3 gPB(T / 64, BB * H);
  posbias_k<<<gPB, 128, 0, stream>>>(q_bf, posk_bf, c2p_full, inv_bias_scale);
  posbias_k<<<gPB, 128, 0, stream>>>(k_bf, posq_bf, p2c_full, inv_bias_scale);

  // 5) bucket LUTs
  lut_k<<<8, 256, 0, stream>>>(lutc, lutp);

  // 6) fused attention
  dim3 gAtt(T / 64, BB * H);
  attn_k<<<gAtt, 128, 0, stream>>>(q_bf, k_bf, vT_bf, c2p_full, p2c_full, lutc,
                                   lutp, mask, ctx_bf);

  // 7) output projection -> f32 d_out
  gemm_bf16_k<<<gBig, 128, 0, stream>>>(ctx_bf, DD, WoT, DD, bo, d_out, DD, DD, 1);
  (void)in_sizes; (void)n_in; (void)out_size; (void)ws_size;
}
